// TransNetSplit2_72421738545468
// MI455X (gfx1250) — compile-verified, matrix operand order checked by reading, not run on hardware
//
#include <hip/hip_runtime.h>
#include <hip/hip_bf16.h>
#include <math.h>

typedef __attribute__((ext_vector_type(16))) _Float16 v16h;
typedef __attribute__((ext_vector_type(8)))  _Float16 v8h;
typedef __attribute__((ext_vector_type(8)))  float    v8f;

#define DT_C 0.1f
#define TT_C 10.0f   /* t = DT/EPS */

// ---------------- CDNA5 async-copy helpers (inline asm; no builtin on this toolchain) ----

// GLOBAL_LOAD_ASYNC_TO_LDS_B128: 16B per lane, memory -> LDS, tracked by ASYNCcnt.
__device__ __forceinline__ void async_copy_b128(unsigned lds_addr, const _Float16* gaddr) {
  asm volatile("global_load_async_to_lds_b128 %0, %1, off"
               :: "v"(lds_addr), "v"(gaddr)
               : "memory");
}
__device__ __forceinline__ void wait_async0() {
  asm volatile("s_wait_asynccnt 0x0" ::: "memory");
}
// Low 32 bits of the flat address of a __shared__ object = allocation-relative LDS byte offset.
__device__ __forceinline__ unsigned lds_off(const void* p) {
  return (unsigned)(unsigned long long)p;
}

// ---------------- helper kernels ----------------

__global__ void cvt_f32_to_f16(const float* __restrict__ s, _Float16* __restrict__ d, int n) {
  int i = blockIdx.x * blockDim.x + threadIdx.x;
  if (i < n) d[i] = (_Float16)s[i];
}

// Build A = I + E, X0 = I - E (Newton seed), Kt = E + t*J (J = diag(0,I)) for one layer.
// E = DT * [[0, W], [-W^T, 0]]   (512x512, W is 256x256 row-major)
__global__ void build_mats(const float* __restrict__ W, float* __restrict__ A,
                           float* __restrict__ X0, float* __restrict__ Kt) {
  int idx = blockIdx.x * blockDim.x + threadIdx.x;   // 512*512 elements
  int i = idx >> 9, j = idx & 511;
  float e = 0.f;
  if (i < 256 && j >= 256)      e =  DT_C * W[i * 256 + (j - 256)];
  else if (i >= 256 && j < 256) e = -DT_C * W[j * 256 + (i - 256)];
  float id = (i == j) ? 1.f : 0.f;
  A[idx]  = id + e;
  X0[idx] = id - e;
  Kt[idx] = e + ((i == j && i >= 256) ? TT_C : 0.f);
}

// C = alpha*A@B + beta*Cin, all 512x512 fp32 row-major (precompute path; exactness matters)
__global__ __launch_bounds__(256) void gemm512_f32(float* __restrict__ C,
                                                   const float* __restrict__ A,
                                                   const float* __restrict__ Bm,
                                                   float alpha, float beta,
                                                   const float* __restrict__ Cin) {
  const int N = 512;
  __shared__ float As[64][17];
  __shared__ float Bs[16][65];
  int tx = threadIdx.x & 15, ty = threadIdx.x >> 4;
  int m0 = blockIdx.x * 64, n0 = blockIdx.y * 64;
  float acc[4][4] = {};
  for (int k0 = 0; k0 < N; k0 += 16) {
    #pragma unroll
    for (int r = 0; r < 4; ++r) {
      int idx = threadIdx.x + r * 256;
      As[idx >> 4][idx & 15] = A[(m0 + (idx >> 4)) * N + k0 + (idx & 15)];
    }
    #pragma unroll
    for (int r = 0; r < 4; ++r) {
      int idx = threadIdx.x + r * 256;
      Bs[idx >> 6][idx & 63] = Bm[(k0 + (idx >> 6)) * N + n0 + (idx & 63)];
    }
    __syncthreads();
    #pragma unroll
    for (int kk = 0; kk < 16; ++kk) {
      float av[4], bv[4];
      #pragma unroll
      for (int i = 0; i < 4; ++i) av[i] = As[ty * 4 + i][kk];
      #pragma unroll
      for (int j = 0; j < 4; ++j) bv[j] = Bs[kk][tx * 4 + j];
      #pragma unroll
      for (int i = 0; i < 4; ++i)
        #pragma unroll
        for (int j = 0; j < 4; ++j) acc[i][j] += av[i] * bv[j];
    }
    __syncthreads();
  }
  #pragma unroll
  for (int i = 0; i < 4; ++i)
    #pragma unroll
    for (int j = 0; j < 4; ++j) {
      int r = m0 + ty * 4 + i, c = n0 + tx * 4 + j;
      float v = alpha * acc[i][j];
      if (beta != 0.f) v += beta * Cin[r * N + c];
      C[r * N + c] = v;
    }
}

// RHS for a trans layer: r1 = u + DT*b ; r2 = (v + t*tanh(u)) / (1+t). f16 output.
__global__ void prep_rhs(const float* __restrict__ z, const float* __restrict__ b,
                         _Float16* __restrict__ rhs) {
  int i = blockIdx.x * blockDim.x + threadIdx.x;   // over B*512
  int col = i & 511;
  int row = i >> 9;
  float zv = z[i];
  float v;
  if (col < 256) v = zv + DT_C * b[col];
  else           v = (zv + TT_C * tanhf(z[(row << 9) + col - 256])) * (1.0f / (1.0f + TT_C));
  rhs[i] = (_Float16)v;
}

// ---------------- main WMMA GEMM (async double-buffered tiles) ----------------
// C[i,n] = sum_k Ah[i,k] * Bth[n,k]  (Bt = B^T, both f16 row-major, ld = K), f32 accumulate.
// MODE 0 (input layer, N=256): v = acc + bias[n]; z_out[i,n] = v; z_out[i,256+n] = tanh(v)
// MODE 1 (trans layer, N=512): n<256:  z_out = acc - z_in[i,n]     - DT*bias[n]
//                              n>=256: z_out = acc - z_in[i,n] - t*tanh(z_in[i,n-256])
template <int MODE, int K>
__global__ __launch_bounds__(256) void gemm_wmma(const _Float16* __restrict__ Ah,
                                                 const _Float16* __restrict__ Bth,
                                                 const float* __restrict__ bias,
                                                 const float* __restrict__ z_in,
                                                 float* __restrict__ z_out) {
  __shared__ _Float16 As[2][128 * 40];   // 128 rows x 32 k, padded to 40 halves, double-buffered
  __shared__ _Float16 Bs[2][128 * 40];
  const int tid  = threadIdx.x;
  const int m0   = blockIdx.x * 128;
  const int n0   = blockIdx.y * 128;
  const int lane = tid & 31;
  const int wave = tid >> 5;
  const int wm   = wave & 1;    // 2 waves along M (64 rows each)
  const int wn   = wave >> 1;   // 4 waves along N (32 cols each)
  const int kb   = (lane >> 4) * 8;
  const int mr   = lane & 15;

  // Per-thread staging slots: thread covers rows r0 and r0+64, 8 halves each (16B).
  const int r0 = tid >> 2;             // 0..63
  const int c0 = (tid & 3) * 8;        // 0,8,16,24
  const _Float16* gA0 = Ah  + (size_t)(m0 + r0)      * K + c0;
  const _Float16* gA1 = Ah  + (size_t)(m0 + r0 + 64) * K + c0;
  const _Float16* gB0 = Bth + (size_t)(n0 + r0)      * K + c0;
  const _Float16* gB1 = Bth + (size_t)(n0 + r0 + 64) * K + c0;
  unsigned lA0[2], lA1[2], lB0[2], lB1[2];
  #pragma unroll
  for (int b = 0; b < 2; ++b) {
    lA0[b] = lds_off(&As[b][r0 * 40 + c0]);
    lA1[b] = lds_off(&As[b][(r0 + 64) * 40 + c0]);
    lB0[b] = lds_off(&Bs[b][r0 * 40 + c0]);
    lB1[b] = lds_off(&Bs[b][(r0 + 64) * 40 + c0]);
  }

  v8f acc[4][2] = {};   // [m-subtile][n-subtile], 16x16 each
  union F16x16 { v16h v; v8h h[2]; };

  const int nk = K >> 5;

  // prologue: fill buffer 0
  async_copy_b128(lA0[0], gA0);
  async_copy_b128(lA1[0], gA1);
  async_copy_b128(lB0[0], gB0);
  async_copy_b128(lB1[0], gB1);

  #pragma unroll 2
  for (int i = 0; i < nk; ++i) {
    const int cur = i & 1;
    wait_async0();        // this wave's loads into buf[cur] have landed in LDS
    __syncthreads();      // all waves: buf[cur] ready AND everyone done reading buf[1-cur]

    if (i + 1 < nk) {     // overlap next tile's DMA with this tile's WMMAs
      const int k1 = (i + 1) << 5;
      async_copy_b128(lA0[1 - cur], gA0 + k1);
      async_copy_b128(lA1[1 - cur], gA1 + k1);
      async_copy_b128(lB0[1 - cur], gB0 + k1);
      async_copy_b128(lB1[1 - cur], gB1 + k1);
    }

    F16x16 a[4], b[2];
    #pragma unroll
    for (int mt = 0; mt < 4; ++mt) {
      int row = wm * 64 + mt * 16 + mr;
      a[mt].h[0] = *(const v8h*)&As[cur][row * 40 + kb];
      a[mt].h[1] = *(const v8h*)&As[cur][row * 40 + 16 + kb];
    }
    #pragma unroll
    for (int nt = 0; nt < 2; ++nt) {
      int row = wn * 32 + nt * 16 + mr;
      b[nt].h[0] = *(const v8h*)&Bs[cur][row * 40 + kb];
      b[nt].h[1] = *(const v8h*)&Bs[cur][row * 40 + 16 + kb];
    }
    #pragma unroll
    for (int mt = 0; mt < 4; ++mt)
      #pragma unroll
      for (int nt = 0; nt < 2; ++nt)
        acc[mt][nt] = __builtin_amdgcn_wmma_f32_16x16x32_f16(
            false, a[mt].v, false, b[nt].v, (short)0, acc[mt][nt], false, false);
  }

  // epilogue; C layout: lane<16 -> M=r, lane>=16 -> M=8+r; N = lane&15
  const int rbase = (lane >> 4) * 8;
  const int nloc  = lane & 15;
  #pragma unroll
  for (int mt = 0; mt < 4; ++mt)
    #pragma unroll
    for (int nt = 0; nt < 2; ++nt)
      #pragma unroll
      for (int r = 0; r < 8; ++r) {
        int gi = m0 + wm * 64 + mt * 16 + rbase + r;
        int gj = n0 + wn * 32 + nt * 16 + nloc;
        float v = acc[mt][nt][r];
        if (MODE == 0) {
          v += bias[gj];
          z_out[(size_t)gi * 512 + gj]       = v;
          z_out[(size_t)gi * 512 + 256 + gj] = tanhf(v);
        } else {
          if (gj < 256) {
            v = v - z_in[(size_t)gi * 512 + gj] - DT_C * bias[gj];
          } else {
            float u = z_in[(size_t)gi * 512 + gj - 256];
            v = v - z_in[(size_t)gi * 512 + gj] - TT_C * tanhf(u);
          }
          z_out[(size_t)gi * 512 + gj] = v;
        }
      }
}

// ---------------- head: logits = z[:, :256] @ Wo^T + bo ; softmax ----------------
__global__ __launch_bounds__(256) void head_kernel(const float* __restrict__ z,
                                                   const float* __restrict__ Wo,
                                                   const float* __restrict__ bo,
                                                   float* __restrict__ out) {
  __shared__ float WoL[10 * 256];
  __shared__ float boL[10];
  for (int i = threadIdx.x; i < 2560; i += 256) WoL[i] = Wo[i];
  if (threadIdx.x < 10) boL[threadIdx.x] = bo[threadIdx.x];
  __syncthreads();
  int row = blockIdx.x * 256 + threadIdx.x;
  float acc[10];
  #pragma unroll
  for (int j = 0; j < 10; ++j) acc[j] = boL[j];
  const float* zr = z + (size_t)row * 512;
  for (int k = 0; k < 256; ++k) {
    float zv = zr[k];
    #pragma unroll
    for (int j = 0; j < 10; ++j) acc[j] += zv * WoL[j * 256 + k];
  }
  float mx = acc[0];
  #pragma unroll
  for (int j = 1; j < 10; ++j) mx = fmaxf(mx, acc[j]);
  float s = 0.f;
  #pragma unroll
  for (int j = 0; j < 10; ++j) { acc[j] = expf(acc[j] - mx); s += acc[j]; }
  float inv = 1.f / s;
  #pragma unroll
  for (int j = 0; j < 10; ++j) out[(size_t)row * 10 + j] = acc[j] * inv;
}

// ---------------- driver ----------------
extern "C" void kernel_launch(void* const* d_in, const int* in_sizes, int n_in,
                              void* d_out, int out_size, void* d_ws, size_t ws_size,
                              hipStream_t stream) {
  (void)n_in; (void)out_size; (void)ws_size;
  const float* x  = (const float*)d_in[0];
  const float* Wi = (const float*)d_in[1];
  const float* bi = (const float*)d_in[2];
  const float* Wl[4] = {(const float*)d_in[3], (const float*)d_in[5],
                        (const float*)d_in[7], (const float*)d_in[9]};
  const float* bl[4] = {(const float*)d_in[4], (const float*)d_in[6],
                        (const float*)d_in[8], (const float*)d_in[10]};
  const float* Wo = (const float*)d_in[11];
  const float* bo = (const float*)d_in[12];
  float* out = (float*)d_out;
  const int B = in_sizes[0] / 256;   // 32768

  char* ws = (char*)d_ws;
  size_t off = 0;
  auto take = [&](size_t bytes) {
    off = (off + 255) & ~(size_t)255;
    char* p = ws + off;
    off += bytes;
    return p;
  };
  float*    zA    = (float*)take((size_t)B * 512 * 4);
  float*    zB    = (float*)take((size_t)B * 512 * 4);
  _Float16* rhs   = (_Float16*)take((size_t)B * 512 * 2);
  _Float16* xh    = (_Float16*)take((size_t)B * 256 * 2);
  _Float16* Wih   = (_Float16*)take((size_t)256 * 256 * 2);
  _Float16* Ph    = (_Float16*)take((size_t)4 * 512 * 512 * 2);
  float*    matA  = (float*)take((size_t)512 * 512 * 4);
  float*    matX  = (float*)take((size_t)512 * 512 * 4);
  float*    matXn = (float*)take((size_t)512 * 512 * 4);
  float*    matT  = (float*)take((size_t)512 * 512 * 4);
  float*    matKt = (float*)take((size_t)512 * 512 * 4);

  // --- precompute per-layer P^T = K^T A^{-1} (fp32, Newton iteration), cast to f16 ---
  for (int l = 0; l < 4; ++l) {
    build_mats<<<1024, 256, 0, stream>>>(Wl[l], matA, matX, matKt);
    float* xc = matX; float* xn = matXn;
    for (int it = 0; it < 3; ++it) {
      gemm512_f32<<<dim3(8, 8), 256, 0, stream>>>(matT, matA, xc, 1.f, 0.f, matA);
      gemm512_f32<<<dim3(8, 8), 256, 0, stream>>>(xn, xc, matT, -1.f, 2.f, xc);
      float* t = xc; xc = xn; xn = t;
    }
    gemm512_f32<<<dim3(8, 8), 256, 0, stream>>>(matT, matKt, xc, 1.f, 0.f, matA);
    cvt_f32_to_f16<<<(512 * 512) / 256, 256, 0, stream>>>(matT, Ph + (size_t)l * 512 * 512, 512 * 512);
  }

  // --- input layer: z = x @ Wi^T + bi ; z = [z, tanh(z)] ---
  cvt_f32_to_f16<<<(B * 256) / 256, 256, 0, stream>>>(x, xh, B * 256);
  cvt_f32_to_f16<<<(256 * 256) / 256, 256, 0, stream>>>(Wi, Wih, 256 * 256);
  gemm_wmma<0, 256><<<dim3(B / 128, 2), 256, 0, stream>>>(xh, Wih, bi, zA, zA);

  // --- 4 trans layers: one B x 512 x 512 WMMA GEMM each ---
  float* zc = zA; float* zn = zB;
  for (int l = 0; l < 4; ++l) {
    prep_rhs<<<(B * 512) / 256, 256, 0, stream>>>(zc, bl[l], rhs);
    gemm_wmma<1, 512><<<dim3(B / 128, 4), 256, 0, stream>>>(rhs, Ph + (size_t)l * 512 * 512,
                                                            bl[l], zc, zn);
    float* t = zc; zc = zn; zn = t;
  }

  // --- head + softmax ---
  head_kernel<<<B / 256, 256, 0, stream>>>(zc, Wo, bo, out);
}
